// TiSASTransformerEncoderBlock_43550968381453
// MI455X (gfx1250) — compile-verified, matrix operand order checked by reading, not run on hardware
//
#include <hip/hip_runtime.h>
#include <hip/hip_bf16.h>
#include <math.h>

// ---------------------------------------------------------------------------
// TiSAS transformer encoder block for MI455X (gfx1250, wave32, WMMA).
// B=4, S=512, D=128, H=4, HD=32.  Bandwidth-bound on time_matrix_{K,V} (1 GB).
// ---------------------------------------------------------------------------

#define BATCH 4
#define SEQ   512
#define DMODEL 128
#define NHEAD 4
#define HDIM  32
#define NROW  (BATCH * SEQ)           // 2048
#define NEGV  (-4294967295.0f)        // -2^32 + 1

typedef _Float16 half_t;
typedef __attribute__((ext_vector_type(16))) _Float16 v16h;
typedef __attribute__((ext_vector_type(8)))  _Float16 v8h;
typedef __attribute__((ext_vector_type(8)))  float    v8f;
typedef __attribute__((ext_vector_type(4)))  float    v4f;

// LDS row strides (halves), padded for bank spread; multiples of 8 keep the
// two 16-byte fragment runs aligned.
#define SA_STRIDE  136   // GEMM A rows (16 x 128)
#define SW_STRIDE  136   // GEMM W^T rows (64 x 128)
#define KH_STRIDE  40    // attention k rows (512 x 32)
#define VT_STRIDE  520   // attention v^T rows (32 x 512)
#define AH_STRIDE  520   // attention attn-f16 rows (16 x 512)
#define QH_STRIDE  40    // attention q rows (16 x 32)

// Fragment loader: 16-bit WMMA operand for a 32-wide K chunk stored
// K-contiguous at `base`.  Lane half `hs` needs K = {8hs..8hs+7} and
// {16+8hs..16+8hs+7}: exactly two aligned 16-byte LDS reads (ds_load_b128).
__device__ __forceinline__ v16h load_frag(const half_t* base, int hs) {
    const v8h lo = *(const v8h*)(base + 8 * hs);
    const v8h hi = *(const v8h*)(base + 16 + 8 * hs);
    v16h r;
    #pragma unroll
    for (int j = 0; j < 8; ++j) { r[j] = lo[j]; r[8 + j] = hi[j]; }
    return r;
}

__device__ __forceinline__ v8f wmma_f32_f16(v16h a, v16h b, v8f c) {
    return __builtin_amdgcn_wmma_f32_16x16x32_f16(false, a, false, b,
                                                  (short)0, c, false, false);
}

// ---------------------------------------------------------------------------
// LayerNorm over D=128 (biased variance, eps=1e-8).  One block per row.
// ---------------------------------------------------------------------------
__global__ __launch_bounds__(128)
void ln_kernel(const float* __restrict__ in1, const float* __restrict__ g,
               const float* __restrict__ beta, float* __restrict__ out) {
    int row = blockIdx.x;
    int t = threadIdx.x;
    float v = in1[(size_t)row * DMODEL + t];
    float s = v, s2 = v * v;
    #pragma unroll
    for (int o = 1; o < 32; o <<= 1) {
        s  += __shfl_xor(s, o, 32);
        s2 += __shfl_xor(s2, o, 32);
    }
    __shared__ float ws[4], ws2[4];
    int w = t >> 5, l = t & 31;
    if (l == 0) { ws[w] = s; ws2[w] = s2; }
    __syncthreads();
    float S1 = ws[0] + ws[1] + ws[2] + ws[3];
    float S2 = ws2[0] + ws2[1] + ws2[2] + ws2[3];
    float mean = S1 * (1.0f / DMODEL);
    float var  = S2 * (1.0f / DMODEL) - mean * mean;
    float r = rsqrtf(var + 1e-8f);
    out[(size_t)row * DMODEL + t] = (v - mean) * r * g[t] + beta[t];
}

// ---------------------------------------------------------------------------
// C = epi(A(2048x128) @ W(128x128) + bias [+ resid]).
// Block = 128 threads (4 waves), block tile 16x64, wave tile 16x16.
// W staged transposed (K-contiguous) so both fragments are 2x ds_load_b128.
// Dual accumulators break the WMMA->WMMA RAW hazard chain.
// ---------------------------------------------------------------------------
__global__ __launch_bounds__(128)
void gemm128_kernel(const float* __restrict__ A, const float* __restrict__ W,
                    const float* __restrict__ bias, const float* __restrict__ resid,
                    float* __restrict__ out, int epilogue) {
    __shared__ half_t sA[16 * SA_STRIDE];    // [m][k]
    __shared__ half_t sWt[64 * SW_STRIDE];   // [n][k]  (transposed)
    int tid = threadIdx.x, lane = tid & 31, wave = tid >> 5;
    int row0 = blockIdx.x * 16;
    int col0 = blockIdx.y * 64;

    // Stage A: 16x128 floats as f16, float4 global loads.
    for (int i = tid; i < (16 * DMODEL) / 4; i += 128) {
        int r = i >> 5, c = (i & 31) * 4;
        v4f t = *(const v4f*)(A + (size_t)(row0 + r) * DMODEL + c);
        sA[r * SA_STRIDE + c + 0] = (half_t)t.x;
        sA[r * SA_STRIDE + c + 1] = (half_t)t.y;
        sA[r * SA_STRIDE + c + 2] = (half_t)t.z;
        sA[r * SA_STRIDE + c + 3] = (half_t)t.w;
    }
    // Stage W transposed: read [k][n] coalesced, write [n][k].
    for (int i = tid; i < (DMODEL * 64) / 4; i += 128) {
        int r = i >> 4, c = (i & 15) * 4;    // r = k, c = n_local
        v4f t = *(const v4f*)(W + (size_t)r * DMODEL + col0 + c);
        sWt[(c + 0) * SW_STRIDE + r] = (half_t)t.x;
        sWt[(c + 1) * SW_STRIDE + r] = (half_t)t.y;
        sWt[(c + 2) * SW_STRIDE + r] = (half_t)t.z;
        sWt[(c + 3) * SW_STRIDE + r] = (half_t)t.w;
    }
    __syncthreads();

    int m = lane & 15, hs = lane >> 4, n = lane & 15;
    int nl = wave * 16 + n;
    float bv = bias[col0 + nl];
    v8f acc0 = {bv, bv, bv, bv, bv, bv, bv, bv};
    v8f acc1 = {};

    #pragma unroll
    for (int kk = 0; kk < DMODEL; kk += 64) {
        v16h a0 = load_frag(sA + m * SA_STRIDE + kk, hs);
        v16h b0 = load_frag(sWt + nl * SW_STRIDE + kk, hs);
        acc0 = wmma_f32_f16(a0, b0, acc0);
        v16h a1 = load_frag(sA + m * SA_STRIDE + kk + 32, hs);
        v16h b1 = load_frag(sWt + nl * SW_STRIDE + kk + 32, hs);
        acc1 = wmma_f32_f16(a1, b1, acc1);
    }
    v8f acc = acc0 + acc1;

    #pragma unroll
    for (int r = 0; r < 8; ++r) {
        int grow = row0 + r + 8 * hs;    // D layout: VGPR r -> M = r + 8*half
        int gcol = col0 + nl;
        float val = acc[r];
        if (resid) val += resid[(size_t)grow * DMODEL + gcol];
        if (epilogue == 1)
            val = 0.5f * val * (1.0f + erff(val * 0.70710678118654752f));
        out[(size_t)grow * DMODEL + gcol] = val;
    }
}

// ---------------------------------------------------------------------------
// Fused attention: one block per (b, h, 16 q-rows).  256 threads = 8 waves.
// Dynamic LDS layout (bytes):
//   energy  16*512 f32                    @ 0       (32768)
//   kh      512 x KH_STRIDE f16           @ 32768   (40960)
//   vhT      32 x VT_STRIDE f16           @ 73728   (33280)
//   attn_h   16 x AH_STRIDE f16           @ 107008  (16640)
//   qf      16*32 f32                     @ 123648  ( 2048)
//   qh       16 x QH_STRIDE f16           @ 125696  ( 1280)
//   obuf    16*32 f32                     @ 126976  ( 2048)   total 129024
// ---------------------------------------------------------------------------
#define SMEM_TOTAL 129024

__global__ __launch_bounds__(256)
void attn_kernel(const float* __restrict__ qb, const float* __restrict__ kb,
                 const float* __restrict__ vb, const float* __restrict__ tmK,
                 const float* __restrict__ tmV, const float* __restrict__ amask,
                 const unsigned char* __restrict__ pmask, float* __restrict__ outO) {
    extern __shared__ char smem[];
    float*  energy = (float*)smem;
    half_t* kh     = (half_t*)(smem + 32768);
    half_t* vhT    = (half_t*)(smem + 73728);
    half_t* attn_h = (half_t*)(smem + 107008);
    float*  qf     = (float*)(smem + 123648);
    half_t* qh     = (half_t*)(smem + 125696);
    float*  obuf   = (float*)(smem + 126976);

    int bid = blockIdx.x;
    int qt = bid & 31;
    int bh = bid >> 5;
    int h = bh & (NHEAD - 1);
    int b = bh >> 2;
    int q0 = qt * 16;
    int tid = threadIdx.x, lane = tid & 31, wave = tid >> 5;

    const size_t baseKV = ((size_t)b * SEQ) * DMODEL + h * HDIM;  // + s*128 + d

    // Stage k (row-major, K=d contiguous) and v transposed (K=kcol contiguous).
    for (int i = tid; i < (SEQ * HDIM) / 4; i += 256) {
        int kk = i >> 3, d0 = (i & 7) * 4;
        const size_t g = baseKV + (size_t)kk * DMODEL + d0;
        v4f tk = *(const v4f*)(kb + g);
        v4f tv = *(const v4f*)(vb + g);
        kh[kk * KH_STRIDE + d0 + 0] = (half_t)tk.x;
        kh[kk * KH_STRIDE + d0 + 1] = (half_t)tk.y;
        kh[kk * KH_STRIDE + d0 + 2] = (half_t)tk.z;
        kh[kk * KH_STRIDE + d0 + 3] = (half_t)tk.w;
        vhT[(d0 + 0) * VT_STRIDE + kk] = (half_t)tv.x;
        vhT[(d0 + 1) * VT_STRIDE + kk] = (half_t)tv.y;
        vhT[(d0 + 2) * VT_STRIDE + kk] = (half_t)tv.z;
        vhT[(d0 + 3) * VT_STRIDE + kk] = (half_t)tv.w;
    }
    for (int i = tid; i < (16 * HDIM) / 4; i += 256) {
        int r = i >> 3, d0 = (i & 7) * 4;
        v4f t = *(const v4f*)(qb + baseKV + (size_t)(q0 + r) * DMODEL + d0);
        qf[r * HDIM + d0 + 0] = t.x;  qh[r * QH_STRIDE + d0 + 0] = (half_t)t.x;
        qf[r * HDIM + d0 + 1] = t.y;  qh[r * QH_STRIDE + d0 + 1] = (half_t)t.y;
        qf[r * HDIM + d0 + 2] = t.z;  qh[r * QH_STRIDE + d0 + 2] = (half_t)t.z;
        qf[r * HDIM + d0 + 3] = t.w;  qh[r * QH_STRIDE + d0 + 3] = (half_t)t.w;
    }
    __syncthreads();

    int m = lane & 15, hs = lane >> 4, n = lane & 15;

    // ---- Phase 1: energy = q @ k^T  (WMMA, K = HD = 32) --------------------
    {
        v16h aq = load_frag(qh + m * QH_STRIDE, hs);
        for (int kt = wave * 4; kt < wave * 4 + 4; ++kt) {
            v16h bk = load_frag(kh + (kt * 16 + n) * KH_STRIDE, hs);
            v8f c = {};
            c = wmma_f32_f16(aq, bk, c);
            #pragma unroll
            for (int r = 0; r < 8; ++r)
                energy[(r + 8 * hs) * SEQ + kt * 16 + n] = c[r];
        }
    }
    __syncthreads();

    // ---- Phase 2: energy += sum_d tmK[b,q,k,h,d] * q[d]  (stream, NT) ------
    {
        const size_t tmBase = (((size_t)b * SEQ + q0) * SEQ) * DMODEL + h * HDIM;
        int sub = lane >> 3, d0 = (lane & 7) * 4;   // 8 lanes per (q,k) pair
        for (int p = wave * 1024; p < wave * 1024 + 1024; p += 4) {
            int pp = p + sub;
            int qr = pp >> 9, kc = pp & 511;
            const float* src = tmK + tmBase + ((size_t)qr * SEQ + kc) * DMODEL + d0;
            __builtin_prefetch(src + 4 * DMODEL, 0, 0);
            v4f t = __builtin_nontemporal_load((const v4f*)src);
            float dp = t.x * qf[qr * HDIM + d0]     + t.y * qf[qr * HDIM + d0 + 1]
                     + t.z * qf[qr * HDIM + d0 + 2] + t.w * qf[qr * HDIM + d0 + 3];
            dp += __shfl_xor(dp, 1, 32);
            dp += __shfl_xor(dp, 2, 32);
            dp += __shfl_xor(dp, 4, 32);
            if ((lane & 7) == 0) energy[qr * SEQ + kc] += dp;
        }
    }
    __syncthreads();

    // ---- Phase 3: scale + mask + softmax (one wave per 2 rows) -------------
    {
        const float inv_scale = 0.17677669529663687f;  // 1/sqrt(32)
        for (int q = wave * 2; q < wave * 2 + 2; ++q) {
            int qg = q0 + q;
            float mx = -3.4e38f;
            for (int kk = lane; kk < SEQ; kk += 32) {
                float e = energy[q * SEQ + kk] * inv_scale + amask[(size_t)qg * SEQ + kk];
                if (pmask[b * SEQ + kk]) e = NEGV;
                energy[q * SEQ + kk] = e;
                mx = fmaxf(mx, e);
            }
            #pragma unroll
            for (int o = 1; o < 32; o <<= 1) mx = fmaxf(mx, __shfl_xor(mx, o, 32));
            float sum = 0.0f;
            for (int kk = lane; kk < SEQ; kk += 32) {
                float ex = __expf(energy[q * SEQ + kk] - mx);
                energy[q * SEQ + kk] = ex;
                sum += ex;
            }
            #pragma unroll
            for (int o = 1; o < 32; o <<= 1) sum += __shfl_xor(sum, o, 32);
            float inv = 1.0f / sum;
            for (int kk = lane; kk < SEQ; kk += 32) {
                float a = energy[q * SEQ + kk] * inv;
                energy[q * SEQ + kk] = a;           // f32 copy for tmV stream
                attn_h[q * AH_STRIDE + kk] = (half_t)a;  // f16 copy for WMMA
            }
        }
    }
    __syncthreads();

    // ---- Phase 4a: O = attn @ v  (WMMA, waves 0..1, dual accumulators) -----
    if (wave < 2) {
        v8f c0 = {}, c1 = {};
        for (int kk = 0; kk < SEQ; kk += 64) {
            v16h a0 = load_frag(attn_h + m * AH_STRIDE + kk, hs);
            v16h b0 = load_frag(vhT + (wave * 16 + n) * VT_STRIDE + kk, hs);
            c0 = wmma_f32_f16(a0, b0, c0);
            v16h a1 = load_frag(attn_h + m * AH_STRIDE + kk + 32, hs);
            v16h b1 = load_frag(vhT + (wave * 16 + n) * VT_STRIDE + kk + 32, hs);
            c1 = wmma_f32_f16(a1, b1, c1);
        }
        v8f c = c0 + c1;
        #pragma unroll
        for (int r = 0; r < 8; ++r)
            obuf[(r + 8 * hs) * HDIM + wave * 16 + n] = c[r];
    }

    // ---- Phase 4b: O += sum_k attn * tmV  (stream, NT, lane = d) -----------
    float acc0 = 0.0f, acc1 = 0.0f;
    {
        int q = wave * 2;
        size_t tBase = (((size_t)b * SEQ + q0 + q) * SEQ) * DMODEL + h * HDIM + lane;
        const float* a0 = energy + q * SEQ;
        const float* a1 = energy + (q + 1) * SEQ;
        const size_t qstride = (size_t)SEQ * DMODEL;
        #pragma unroll 4
        for (int k = 0; k < SEQ; ++k) {
            size_t off = tBase + (size_t)k * DMODEL;
            __builtin_prefetch(tmV + off + 8 * DMODEL, 0, 0);
            acc0 += a0[k] * __builtin_nontemporal_load(tmV + off);
            acc1 += a1[k] * __builtin_nontemporal_load(tmV + off + qstride);
        }
    }
    __syncthreads();   // obuf base tiles from phase 4a are complete
    obuf[(wave * 2) * HDIM + lane]     += acc0;
    obuf[(wave * 2 + 1) * HDIM + lane] += acc1;
    __syncthreads();

    // ---- Write O tile back (layout [b][s][h*HD + d]) -----------------------
    for (int i = tid; i < 16 * HDIM; i += 256) {
        int r = i >> 5, d = i & 31;
        outO[baseKV + (size_t)(q0 + r) * DMODEL + d] = obuf[i];
    }
}

// ---------------------------------------------------------------------------
extern "C" void kernel_launch(void* const* d_in, const int* in_sizes, int n_in,
                              void* d_out, int out_size, void* d_ws, size_t ws_size,
                              hipStream_t stream) {
    (void)in_sizes; (void)n_in; (void)out_size; (void)ws_size;

    const float* embed = (const float*)d_in[0];
    const float* tmK   = (const float*)d_in[1];
    const float* tmV   = (const float*)d_in[2];
    const float* amask = (const float*)d_in[3];
    const unsigned char* pmask = (const unsigned char*)d_in[4];
    const float* wq = (const float*)d_in[5];
    const float* bq = (const float*)d_in[6];
    const float* wk = (const float*)d_in[7];
    const float* bk = (const float*)d_in[8];
    const float* wvp = (const float*)d_in[9];
    const float* bvp = (const float*)d_in[10];
    const float* wo = (const float*)d_in[11];
    const float* bo = (const float*)d_in[12];
    const float* ln1g = (const float*)d_in[13];
    const float* ln1b = (const float*)d_in[14];
    const float* ln2g = (const float*)d_in[15];
    const float* ln2b = (const float*)d_in[16];
    const float* w1 = (const float*)d_in[17];
    const float* b1 = (const float*)d_in[18];
    const float* w2 = (const float*)d_in[19];
    const float* b2 = (const float*)d_in[20];

    float* out = (float*)d_out;
    float* ws  = (float*)d_ws;
    const size_t RD = (size_t)NROW * DMODEL;  // 262144 floats per activation

    float* Qn     = ws + 0 * RD;
    float* qbuf   = ws + 1 * RD;
    float* kbuf   = ws + 2 * RD;
    float* vbuf   = ws + 3 * RD;
    float* attn_o = ws + 4 * RD;
    float* xpre   = ws + 5 * RD;
    float* xb     = ws + 6 * RD;
    float* hb     = ws + 7 * RD;

    dim3 gg(NROW / 16, DMODEL / 64);   // (128, 2)

    // 1) Pre-LN on Q path
    ln_kernel<<<NROW, 128, 0, stream>>>(embed, ln1g, ln1b, Qn);
    // 2) q/k/v projections
    gemm128_kernel<<<gg, 128, 0, stream>>>(Qn,    wq,  bq,  nullptr, qbuf, 0);
    gemm128_kernel<<<gg, 128, 0, stream>>>(embed, wk,  bk,  nullptr, kbuf, 0);
    gemm128_kernel<<<gg, 128, 0, stream>>>(embed, wvp, bvp, nullptr, vbuf, 0);
    // 3) fused attention (streams tmK/tmV exactly once, non-temporal)
    attn_kernel<<<BATCH * NHEAD * (SEQ / 16), 256, SMEM_TOTAL, stream>>>(
        qbuf, kbuf, vbuf, tmK, tmV, amask, pmask, attn_o);
    // 4) output projection + residual (x = Qn + O)
    gemm128_kernel<<<gg, 128, 0, stream>>>(attn_o, wo, bo, Qn, xpre, 0);
    // 5) LN2
    ln_kernel<<<NROW, 128, 0, stream>>>(xpre, ln2g, ln2b, xb);
    // 6) FFN: gelu(x@w1+b1) @ w2 + b2 + x
    gemm128_kernel<<<gg, 128, 0, stream>>>(xb, w1, b1, nullptr, hb, 1);
    gemm128_kernel<<<gg, 128, 0, stream>>>(hb, w2, b2, xb, out, 0);
}
